// LinearAttention_88381837017536
// MI455X (gfx1250) — compile-verified, matrix operand order checked by reading, not run on hardware
//
#include <hip/hip_runtime.h>
#include <cstddef>

typedef __attribute__((ext_vector_type(16))) __bf16 v16bf;
typedef __attribute__((ext_vector_type(8)))  float  v8f;

constexpr int kE    = 1024;   // embed dim == K == N for all GEMMs
constexpr int kRows = 2048;   // b*s
constexpr int kBK   = 32;     // K-tile
constexpr int kLDT  = 40;     // padded LDS row stride (elements) = 32 + 8
constexpr int kBUF  = 128 * kLDT;   // one tile buffer (elements)

union Frag {
    v16bf v;
    uint4 q[2];
};

__device__ __forceinline__ unsigned short f2bf(float f) {
    union { float f; unsigned u; } c; c.f = f;
    unsigned r = c.u + 0x7FFFu + ((c.u >> 16) & 1u);   // round-to-nearest-even
    return (unsigned short)(r >> 16);
}
__device__ __forceinline__ float bf2f(unsigned short h) {
    union { unsigned u; float f; } c; c.u = ((unsigned)h) << 16;
    return c.f;
}

// ---------------------------------------------------------------- convert
__global__ __launch_bounds__(256)
void cvt_f32_bf16(const float* __restrict__ src, unsigned short* __restrict__ dst, int n) {
    int i = (blockIdx.x * 256 + threadIdx.x) * 4;
    if (i + 3 < n) {
        float4 f = *reinterpret_cast<const float4*>(src + i);
        ushort4 o;
        o.x = f2bf(f.x); o.y = f2bf(f.y); o.z = f2bf(f.z); o.w = f2bf(f.w);
        *reinterpret_cast<ushort4*>(dst + i) = o;
    }
}

// ------------------------------------------------- shared WMMA GEMM core
// C(2048x1024) = A(2048x1024,bf16) * B^T where B is (1024x1024,bf16) row-major
// (both operands K-contiguous: "NT" layout). Block tile 128x128, 8 waves,
// double-buffered LDS (one barrier per K-step), prefetch 2 tiles ahead.
__device__ __forceinline__ void gemm_core(const unsigned short* __restrict__ A,
                                          const unsigned short* __restrict__ B,
                                          unsigned short* smA, unsigned short* smB,
                                          v8f acc[2][4]) {
    const int t     = threadIdx.x;
    const int lane  = t & 31;
    const int wave  = t >> 5;         // 0..7
    const int waveM = wave >> 1;      // 0..3  -> 32 rows each
    const int waveN = wave & 1;       // 0..1  -> 64 cols each
    const int lm    = lane & 15;
    const int lh    = lane >> 4;

    // tile staging: thread -> (row 0..127, 16-element half of a 32-wide row)
    const int ldRow  = t >> 1;
    const int ldHalf = (t & 1) * 16;
    const int sOff   = ldRow * kLDT + ldHalf;

    const unsigned short* gA = A + (size_t)(blockIdx.x * 128 + ldRow) * kE + ldHalf;
    const unsigned short* gB = B + (size_t)(blockIdx.y * 128 + ldRow) * kE + ldHalf;

    // fragment base addresses (per sub-tile row, per lane)
    const int faRow0 = (waveM * 32 + lm) * kLDT + lh * 8;
    const int fbRow0 = (waveN * 64 + lm) * kLDT + lh * 8;

    // ---- stage tile 0 into buffer 0
    {
        uint4 a0 = reinterpret_cast<const uint4*>(gA)[0];
        uint4 a1 = reinterpret_cast<const uint4*>(gA)[1];
        uint4 b0 = reinterpret_cast<const uint4*>(gB)[0];
        uint4 b1 = reinterpret_cast<const uint4*>(gB)[1];
        reinterpret_cast<uint4*>(smA + sOff)[0] = a0;
        reinterpret_cast<uint4*>(smA + sOff)[1] = a1;
        reinterpret_cast<uint4*>(smB + sOff)[0] = b0;
        reinterpret_cast<uint4*>(smB + sOff)[1] = b1;
    }
    __syncthreads();

    constexpr int NT = kE / kBK;   // 32 K-steps
    for (int kt = 0; kt < NT; ++kt) {
        const int cur = (kt & 1) * kBUF;
        const int nxt = ((kt + 1) & 1) * kBUF;

        // ---- stage tile kt+1 into the other buffer (overlaps WMMAs below)
        if (kt + 1 < NT) {
            const unsigned short* nA = gA + (kt + 1) * kBK;
            const unsigned short* nB = gB + (kt + 1) * kBK;
            uint4 a0 = reinterpret_cast<const uint4*>(nA)[0];
            uint4 a1 = reinterpret_cast<const uint4*>(nA)[1];
            uint4 b0 = reinterpret_cast<const uint4*>(nB)[0];
            uint4 b1 = reinterpret_cast<const uint4*>(nB)[1];
            reinterpret_cast<uint4*>(smA + nxt + sOff)[0] = a0;
            reinterpret_cast<uint4*>(smA + nxt + sOff)[1] = a1;
            reinterpret_cast<uint4*>(smB + nxt + sOff)[0] = b0;
            reinterpret_cast<uint4*>(smB + nxt + sOff)[1] = b1;
            if (kt + 2 < NT) {
                __builtin_prefetch(gA + (kt + 2) * kBK, 0, 3);  // global_prefetch_b8
                __builtin_prefetch(gB + (kt + 2) * kBK, 0, 3);
            }
        }

        // ---- compute from current buffer
        // A fragment (16x32 bf16): lane lm = M row; lh selects K chunk.
        // VGPR pairs hold contiguous K: chunk0 = K[lh*8..+7], chunk1 = K[16+lh*8..+7]
        Frag fa[2];
#pragma unroll
        for (int mi = 0; mi < 2; ++mi) {
            const uint4* p = reinterpret_cast<const uint4*>(
                smA + cur + faRow0 + mi * 16 * kLDT);
            fa[mi].q[0] = p[0];
            fa[mi].q[1] = p[2];          // +32 bytes = +16 elements
        }
#pragma unroll
        for (int ni = 0; ni < 4; ++ni) {
            Frag fb;
            const uint4* p = reinterpret_cast<const uint4*>(
                smB + cur + fbRow0 + ni * 16 * kLDT);
            fb.q[0] = p[0];
            fb.q[1] = p[2];
#pragma unroll
            for (int mi = 0; mi < 2; ++mi) {
                acc[mi][ni] = __builtin_amdgcn_wmma_f32_16x16x32_bf16(
                    false, fa[mi].v, false, fb.v, (short)0, acc[mi][ni], false, false);
            }
        }
        __syncthreads();   // retires reads of cur, publishes nxt
    }
}

// ------------------------------------------------ QKV projections (+ phi)
__global__ __launch_bounds__(256)
void gemm_qkv(const unsigned short* __restrict__ xb,
              const unsigned short* __restrict__ wqb,
              const unsigned short* __restrict__ wkb,
              const unsigned short* __restrict__ wvb,
              unsigned short* __restrict__ qp,
              unsigned short* __restrict__ kp,
              unsigned short* __restrict__ vp) {
    __shared__ unsigned short smA[2 * kBUF];
    __shared__ unsigned short smB[2 * kBUF];
    const int z = blockIdx.z;
    const unsigned short* B = (z == 0) ? wqb : (z == 1) ? wkb : wvb;
    unsigned short*       O = (z == 0) ? qp  : (z == 1) ? kp  : vp;

    v8f acc[2][4] = {};
    gemm_core(xb, B, smA, smB, acc);

    const int t = threadIdx.x, lane = t & 31, wave = t >> 5;
    const int waveM = wave >> 1, waveN = wave & 1;
    const int lm = lane & 15, lh = lane >> 4;
    const int rBase = blockIdx.x * 128 + waveM * 32 + lh * 8;
    const int cBase = blockIdx.y * 128 + waveN * 64;
    const bool act = (z != 2);
#pragma unroll
    for (int mi = 0; mi < 2; ++mi)
#pragma unroll
        for (int ni = 0; ni < 4; ++ni)
#pragma unroll
            for (int r = 0; r < 8; ++r) {
                float vr = acc[mi][ni][r];
                if (act) vr = (vr > 0.0f) ? (vr + 1.0f) : __expf(vr);  // elu(x)+1
                O[(size_t)(rBase + mi * 16 + r) * kE + (cBase + ni * 16 + lm)] = f2bf(vr);
            }
}

// -------------------------------------------- causal linear-attention scan
// One workgroup per (b,h). S (64x64) and running k-sum z live in LDS.
__global__ __launch_bounds__(256)
void linattn_scan(const unsigned short* __restrict__ qp,
                  const unsigned short* __restrict__ kp,
                  const unsigned short* __restrict__ vp,
                  unsigned short* __restrict__ att) {
    __shared__ float S[64 * 64];
    __shared__ float zsum[64];
    __shared__ float ql[64], kl[64], vl[64];
    __shared__ float pnum[4 * 64];

    const int t  = threadIdx.x;
    const int j  = t & 63;     // value/output dim
    const int ig = t >> 6;     // i-chunk (4 x 16)
    const int bh = blockIdx.x;
    const int rowBase = (bh >> 4) * 1024;   // b * s
    const int colBase = (bh & 15) * 64;     // h * d

    for (int idx = t; idx < 64 * 64; idx += 256) S[idx] = 0.0f;
    if (t < 64) zsum[t] = 0.0f;
    __syncthreads();

    for (int tt = 0; tt < 1024; ++tt) {
        const size_t base = (size_t)(rowBase + tt) * kE + colBase;
        if (t < 64) {
            float kq = bf2f(kp[base + t]);
            kl[t] = kq;
            ql[t] = bf2f(qp[base + t]);
            vl[t] = bf2f(vp[base + t]);
            zsum[t] += kq;                  // prefix sum includes current step
        }
        __syncthreads();

        // S += k (x) v (includes current t), then numerator partial
        float vj = vl[j];
        float np = 0.0f;
        const int i0 = ig * 16;
#pragma unroll
        for (int ii = 0; ii < 16; ++ii) {
            int i = i0 + ii;
            float s = S[i * 64 + j] + kl[i] * vj;
            S[i * 64 + j] = s;
            np += ql[i] * s;
        }
        pnum[ig * 64 + j] = np;
        __syncthreads();

        if (t < 64) {
            float num = pnum[t] + pnum[64 + t] + pnum[128 + t] + pnum[192 + t];
            float den = 0.0f;
#pragma unroll 8
            for (int i = 0; i < 64; ++i) den += ql[i] * zsum[i];
            att[base + t] = f2bf(num / (den + 1e-6f));
        }
        __syncthreads();
    }
}

// --------------------------------------------- output projection (+ bias)
__global__ __launch_bounds__(256)
void gemm_out(const unsigned short* __restrict__ attb,
              const unsigned short* __restrict__ wob,
              const float* __restrict__ bo,
              float* __restrict__ out) {
    __shared__ unsigned short smA[2 * kBUF];
    __shared__ unsigned short smB[2 * kBUF];

    v8f acc[2][4] = {};
    gemm_core(attb, wob, smA, smB, acc);

    const int t = threadIdx.x, lane = t & 31, wave = t >> 5;
    const int waveM = wave >> 1, waveN = wave & 1;
    const int lm = lane & 15, lh = lane >> 4;
    const int rBase = blockIdx.x * 128 + waveM * 32 + lh * 8;
    const int cBase = blockIdx.y * 128 + waveN * 64;
#pragma unroll
    for (int mi = 0; mi < 2; ++mi)
#pragma unroll
        for (int ni = 0; ni < 4; ++ni) {
            const int col = cBase + ni * 16 + lm;
            const float bias = bo[col];
#pragma unroll
            for (int r = 0; r < 8; ++r)
                out[(size_t)(rBase + mi * 16 + r) * kE + col] = acc[mi][ni][r] + bias;
        }
}

// ------------------------------------------------------------------ host
extern "C" void kernel_launch(void* const* d_in, const int* in_sizes, int n_in,
                              void* d_out, int out_size, void* d_ws, size_t ws_size,
                              hipStream_t stream) {
    const float* x  = (const float*)d_in[0];
    const float* Wq = (const float*)d_in[1];
    const float* Wk = (const float*)d_in[2];
    const float* Wv = (const float*)d_in[3];
    const float* Wo = (const float*)d_in[4];
    const float* bo = (const float*)d_in[5];

    char* ws = (char*)d_ws;
    const size_t MB = 1ull << 20;
    unsigned short* xb   = (unsigned short*)(ws + 0);        // 4 MB
    unsigned short* wqb  = (unsigned short*)(ws + 4  * MB);  // 2 MB
    unsigned short* wkb  = (unsigned short*)(ws + 6  * MB);  // 2 MB
    unsigned short* wvb  = (unsigned short*)(ws + 8  * MB);  // 2 MB
    unsigned short* wob  = (unsigned short*)(ws + 10 * MB);  // 2 MB
    unsigned short* qp   = (unsigned short*)(ws + 12 * MB);  // 4 MB
    unsigned short* kp   = (unsigned short*)(ws + 16 * MB);  // 4 MB
    unsigned short* vp   = (unsigned short*)(ws + 20 * MB);  // 4 MB
    unsigned short* attb = (unsigned short*)(ws + 24 * MB);  // 4 MB

    const int nX = kRows * kE;   // 2,097,152
    const int nW = kE * kE;      // 1,048,576
    cvt_f32_bf16<<<nX / 1024, 256, 0, stream>>>(x,  xb,  nX);
    cvt_f32_bf16<<<nW / 1024, 256, 0, stream>>>(Wq, wqb, nW);
    cvt_f32_bf16<<<nW / 1024, 256, 0, stream>>>(Wk, wkb, nW);
    cvt_f32_bf16<<<nW / 1024, 256, 0, stream>>>(Wv, wvb, nW);
    cvt_f32_bf16<<<nW / 1024, 256, 0, stream>>>(Wo, wob, nW);

    gemm_qkv<<<dim3(kRows / 128, kE / 128, 3), 256, 0, stream>>>(
        xb, wqb, wkb, wvb, qp, kp, vp);

    linattn_scan<<<32, 256, 0, stream>>>(qp, kp, vp, attb);

    gemm_out<<<dim3(kRows / 128, kE / 128, 1), 256, 0, stream>>>(
        attb, wob, bo, (float*)d_out);
}
